// FullAttention_59167469469977
// MI455X (gfx1250) — compile-verified, matrix operand order checked by reading, not run on hardware
//
#include <hip/hip_runtime.h>
#include <hip/hip_bf16.h>

// ---------------------------------------------------------------------------
// FullAttention fused pipeline for MI455X (gfx1250, wave32, WMMA + TDM).
//   B=8, P=Q=2048, D_IN=H=1024
//   Kernels: cvt (U_w->bf16), trcvt (rep -> repT bf16), proj (WMMA GEMM+ReLU),
//            attn (flash attention, WMMA bf16, fp32 accum).
//   Global->LDS tile staging uses the Tensor Data Mover with DOUBLE-BUFFERED,
//   software-pipelined descriptors (issue chunk c+1, s_wait_tensorcnt(<=2)
//   for chunk c) so the DMA runs underneath the WMMAs.
// ---------------------------------------------------------------------------

typedef __attribute__((ext_vector_type(16))) __bf16 v16bf;
typedef __attribute__((ext_vector_type(8)))  float  v8f;
typedef __attribute__((ext_vector_type(4)))  __bf16 bf4;
typedef __attribute__((ext_vector_type(4)))  unsigned int v4u;
typedef __attribute__((ext_vector_type(8)))  int v8i;
typedef __attribute__((ext_vector_type(4)))  int v4i;

#define BB 8
#define PP 2048
#define QQ 2048
#define DI 1024
#define HH 1024

#if defined(__gfx1250__) && __has_builtin(__builtin_amdgcn_tensor_load_to_lds) && \
    __has_builtin(__builtin_amdgcn_s_wait_tensorcnt)
#define HAVE_TDM 1
#else
#define HAVE_TDM 0
#endif

#if HAVE_TDM
// Low 32 bits of a generic pointer to LDS == LDS byte offset (flat aperture rule).
__device__ inline unsigned lds_off(const void* p) {
  return (unsigned)(unsigned long long)(size_t)p;
}

// 2-D tile DMA: global (row-major, strideD0 elements of 2 bytes) -> LDS,
// with LDS row padding (TDM pad feature).  D# per CDNA5 ISA §8.3-8.6.
// clang-23 builtin form: (v4u g0, v8i g1, v4i g2, v4i g3, v8i g4, i32 cpol)
__device__ inline void tdm_load_2d_bf16(unsigned ldsByteOff, const void* gtile,
                                        unsigned tensorD0, unsigned tensorD1,
                                        unsigned tileD0, unsigned tileD1,
                                        unsigned strideD0,
                                        unsigned padInterval, unsigned padAmount) {
  unsigned long long ga = (unsigned long long)(size_t)gtile;
  v4u g0;
  g0.x = 1u;                                        // count=1, user mode
  g0.y = ldsByteOff;                                // lds_addr [63:32]
  g0.z = (unsigned)ga;                              // global_addr lo
  g0.w = ((unsigned)(ga >> 32) & 0x01FFFFFFu) | (2u << 30);  // addr hi | type=2
  v8i g1;
  g1[0] = (int)((1u << 16)                          // data_size = 2 bytes
                | (1u << 20)                        // pad_enable
                | (padInterval << 22) | (padAmount << 25));
  g1[1] = (int)((tensorD0 & 0xFFFFu) << 16);                        // dim0 lo16
  g1[2] = (int)((tensorD0 >> 16) | ((tensorD1 & 0xFFFFu) << 16));   // dim0 hi/dim1 lo
  g1[3] = (int)((tensorD1 >> 16) | (tileD0 << 16));                 // dim1 hi/tile0
  g1[4] = (int)(tileD1 & 0xFFFFu);                                  // tile1 (tile2=0)
  g1[5] = (int)strideD0;                                            // dim0_stride lo32
  g1[6] = 0;
  g1[7] = 0;
  v4i z4 = {0, 0, 0, 0};                            // 2-D tensor: groups 2/3 unused
  v8i z8 = {0, 0, 0, 0, 0, 0, 0, 0};
  __builtin_amdgcn_tensor_load_to_lds(g0, g1, z4, z4, z8, 0);
}
#endif

// ------------------------------ cvt: f32 -> bf16 ---------------------------
__global__ __launch_bounds__(256) void cvt_kernel(const float* __restrict__ src,
                                                  __bf16* __restrict__ dst, int n) {
  int i = (blockIdx.x * 256 + threadIdx.x) * 4;
  if (i + 3 < n) {
    float4 v = *(const float4*)(src + i);
    bf4 o;
    o.x = (__bf16)v.x; o.y = (__bf16)v.y; o.z = (__bf16)v.z; o.w = (__bf16)v.w;
    *(bf4*)(dst + i) = o;
  }
}

// -------------------- trcvt: rep[B,Q,H] f32 -> repT[B,H,Q] bf16 ------------
__global__ __launch_bounds__(256) void trcvt_kernel(const float* __restrict__ rep,
                                                    __bf16* __restrict__ repT) {
  __shared__ float tile[32][33];
  const int b  = blockIdx.z;
  const int q0 = blockIdx.y * 32;
  const int h0 = blockIdx.x * 32;
  const int tx = threadIdx.x & 31;
  const int ty = threadIdx.x >> 5;            // 0..7
  const float* src = rep + ((size_t)b * QQ + q0) * HH + h0;
#pragma unroll
  for (int j = 0; j < 4; ++j)
    tile[ty + j * 8][tx] = src[(size_t)(ty + j * 8) * HH + tx];
  __syncthreads();
  __bf16* dst = repT + ((size_t)b * HH + h0) * QQ + q0;
#pragma unroll
  for (int j = 0; j < 4; ++j)
    dst[(size_t)(ty + j * 8) * QQ + tx] = (__bf16)tile[tx][ty + j * 8];
}

// -------- proj: Out[r,h] = relu(sum_k X[r,k]*W[h,k]) (* D[h] if useD) ------
// block = 256 thr (8 waves), tile 128(M) x 128(N), wave tile 64x32, K-step 32.
// B tile double-buffered via pipelined TDM; A tile staged manually (needs cvt).
__global__ __launch_bounds__(256) void proj_kernel(const float* __restrict__ X,
                                                   const __bf16* __restrict__ W,
                                                   const float* __restrict__ Dparam,
                                                   __bf16* __restrict__ Out,
                                                   int useD) {
  __shared__ __bf16 sA[128][48];      // 96B stride: 32B-aligned frag rows
  __shared__ __bf16 sB[2][128][48];   // 64B row + 32B pad -> TDM interval=3, amount=7
  const int tid  = threadIdx.x;
  const int wave = tid >> 5, lane = tid & 31;
  const int lh   = lane & 15, ls = lane >> 4;
  const int wm   = wave >> 2;          // 0..1 (M group of 64)
  const int wn   = wave & 3;           // 0..3 (N group of 32)
  const int m0   = blockIdx.y * 128;
  const int n0   = blockIdx.x * 128;

  v8f acc[4][2] = {};

#if HAVE_TDM
  if (wave == 0)     // prefetch B chunk 0
    tdm_load_2d_bf16(lds_off(&sB[0][0][0]), &W[(size_t)n0 * DI],
                     DI, HH, 32, 128, DI, 3, 7);
#endif

  for (int k0 = 0; k0 < DI; k0 += 32) {
    const int buf = (k0 >> 5) & 1;
#if HAVE_TDM
    if (wave == 0) {
      if (k0 + 32 < DI) {
        tdm_load_2d_bf16(lds_off(&sB[buf ^ 1][0][0]), &W[(size_t)n0 * DI + k0 + 32],
                         DI, HH, 32, 128, DI, 3, 7);
        __builtin_amdgcn_s_wait_tensorcnt(1);   // chunk k0 done, k0+32 in flight
      } else {
        __builtin_amdgcn_s_wait_tensorcnt(0);
      }
    }
#else
#pragma unroll
    for (int j = 0; j < 2; ++j) {
      int v = tid + j * 256;
      int row = v >> 2, col = (v & 3) * 8;
      *(uint4*)&sB[buf][row][col] = *(const uint4*)&W[(size_t)(n0 + row) * DI + k0 + col];
    }
#endif
    // stage A (128x32 f32 -> bf16): 1024 float4, 4 per thread (needs conversion);
    // overlaps the in-flight B DMA.
#pragma unroll
    for (int j = 0; j < 4; ++j) {
      int v = tid + j * 256;
      int row = v >> 3, col = (v & 7) * 4;
      float4 x = *(const float4*)&X[(size_t)(m0 + row) * DI + k0 + col];
      bf4 o; o.x = (__bf16)x.x; o.y = (__bf16)x.y; o.z = (__bf16)x.z; o.w = (__bf16)x.w;
      *(bf4*)&sA[row][col] = o;
    }
    __syncthreads();   // tiles ready

    v16bf bfrag[2];
#pragma unroll
    for (int ni = 0; ni < 2; ++ni)
      bfrag[ni] = *(const v16bf*)&sB[buf][wn * 32 + ni * 16 + lh][ls * 16];
#pragma unroll
    for (int mi = 0; mi < 4; ++mi) {
      v16bf afrag = *(const v16bf*)&sA[wm * 64 + mi * 16 + lh][ls * 16];
#pragma unroll
      for (int ni = 0; ni < 2; ++ni)
        acc[mi][ni] = __builtin_amdgcn_wmma_f32_16x16x32_bf16(
            false, afrag, false, bfrag[ni], (short)0, acc[mi][ni], false, false);
    }
    __syncthreads();   // compute done: next iteration may overwrite tiles
  }

  // epilogue: relu (+ D scale), store bf16
#pragma unroll
  for (int ni = 0; ni < 2; ++ni) {
    int col = n0 + wn * 32 + ni * 16 + lh;
    float d = useD ? Dparam[col] : 1.0f;
#pragma unroll
    for (int mi = 0; mi < 4; ++mi) {
      int rbase = m0 + wm * 64 + mi * 16 + ls * 8;   // C frag row = i + 8*ls
#pragma unroll
      for (int i = 0; i < 8; ++i) {
        float v = acc[mi][ni][i];
        v = v > 0.0f ? v : 0.0f;
        Out[(size_t)(rbase + i) * HH + col] = (__bf16)(v * d);
      }
    }
  }
}

// ------------------------------- attn (flash) ------------------------------
// block = 512 thr (16 waves); WG owns 64 P-rows x full H=1024.
// wave (pg,hg): O accum = 16p x 256h = 16 C frags. Per 64-wide Q step:
//   S frag (pg x qg=hg) via K=1024 in 128-chunks, TDM double-buffered/pipelined;
//   S->LDS; online softmax -> P bf16 in LDS; rescale O; PV WMMA vs repT.
__global__ __launch_bounds__(512) void attn_kernel(const __bf16* __restrict__ Up,
                                                   const __bf16* __restrict__ Uq,
                                                   const __bf16* __restrict__ RepT,
                                                   const unsigned char* __restrict__ qmask,
                                                   float* __restrict__ Out) {
  __shared__ __bf16 sUp[2][64][144];  // 256B row + 32B pad -> TDM interval=5, amount=7
  __shared__ __bf16 sUq[2][64][144];
  __shared__ float  sS[64][68];
  __shared__ __bf16 sP[64][80];       // stride 160B (32B-aligned)
  __shared__ float  sM[64], sL[64], sScale[64];

  const int b    = blockIdx.y;
  const int p0   = blockIdx.x * 64;
  const int tid  = threadIdx.x;
  const int wave = tid >> 5, lane = tid & 31;
  const int lh   = lane & 15, ls = lane >> 4;
  const int pg   = wave >> 2;          // 0..3: P-row group (16 rows)
  const int hg   = wave & 3;           // 0..3: H group (256 cols) / S q-group

  const __bf16* upg = Up + ((size_t)b * PP + p0) * HH;
  const __bf16* uqg = Uq + (size_t)b * QQ * HH;
  const __bf16* rtg = RepT + (size_t)b * HH * QQ;
  const unsigned char* qm = qmask + (size_t)b * QQ;

  if (tid < 64) { sM[tid] = -__builtin_inff(); sL[tid] = 0.0f; }

  const int NCH = (QQ / 64) * (HH / 128);   // 256 chunks, 8 per q-step

#if HAVE_TDM
  if (wave == 0) {   // prefetch chunk 0 (q0=0, k0=0)
    tdm_load_2d_bf16(lds_off(&sUp[0][0][0]), &upg[0], HH, PP, 128, 64, HH, 5, 7);
    tdm_load_2d_bf16(lds_off(&sUq[0][0][0]), &uqg[0], HH, QQ, 128, 64, HH, 5, 7);
  }
#endif

  v8f oacc[16] = {};
  int c = 0;   // flat chunk counter

  for (int q0 = 0; q0 < QQ; q0 += 64) {
    v8f sacc = {};
    for (int k0 = 0; k0 < HH; k0 += 128, ++c) {
      const int buf = c & 1;
#if HAVE_TDM
      if (wave == 0) {
        const int cn = c + 1;
        if (cn < NCH) {
          const int qn = (cn >> 3) * 64;     // next chunk's q-step
          const int kn = (cn & 7) * 128;     // next chunk's k offset
          tdm_load_2d_bf16(lds_off(&sUp[cn & 1][0][0]), &upg[kn],
                           HH, PP, 128, 64, HH, 5, 7);
          tdm_load_2d_bf16(lds_off(&sUq[cn & 1][0][0]), &uqg[(size_t)qn * HH + kn],
                           HH, QQ, 128, 64, HH, 5, 7);
          __builtin_amdgcn_s_wait_tensorcnt(2);  // chunk c landed; c+1 in flight
        } else {
          __builtin_amdgcn_s_wait_tensorcnt(0);
        }
      }
      __syncthreads();   // chunk c tiles ready (also orders sM/sL init first time)
#else
      // stage Up & Uq chunks: 64x128 bf16 each = 1024 uint4, 2 per thread
#pragma unroll
      for (int j = 0; j < 2; ++j) {
        int v = tid + j * 512;
        int row = v >> 4, col = (v & 15) * 8;
        *(uint4*)&sUp[buf][row][col] = *(const uint4*)&upg[(size_t)row * HH + k0 + col];
        *(uint4*)&sUq[buf][row][col] =
            *(const uint4*)&uqg[(size_t)(q0 + row) * HH + k0 + col];
      }
      __syncthreads();
#endif
#pragma unroll
      for (int kk = 0; kk < 128; kk += 32) {
        v16bf af = *(const v16bf*)&sUp[buf][pg * 16 + lh][kk + ls * 16];
        v16bf bf = *(const v16bf*)&sUq[buf][hg * 16 + lh][kk + ls * 16];
        sacc = __builtin_amdgcn_wmma_f32_16x16x32_bf16(
            false, af, false, bf, (short)0, sacc, false, false);
      }
      __syncthreads();   // compute done: pipelined DMA may overwrite other buf
    }
    // scatter S fragment to LDS (C layout: row = i + 8*ls, col = lh)
#pragma unroll
    for (int i = 0; i < 8; ++i)
      sS[pg * 16 + i + ls * 8][hg * 16 + lh] = sacc[i];
    __syncthreads();

    // online softmax: one thread per P-row
    if (tid < 64) {
      const int r = tid;
      float mold = sM[r];
      float mt = -__builtin_inff();
#pragma unroll 4
      for (int qq = 0; qq < 64; ++qq)
        if (!qm[q0 + qq]) mt = fmaxf(mt, sS[r][qq]);
      float mnew = fmaxf(mold, mt);
      bool dead = (mnew == -__builtin_inff());   // all masked so far
      float scale = dead ? 1.0f : __expf(mold - mnew);
      float lsum = 0.0f;
#pragma unroll 4
      for (int qq = 0; qq < 64; ++qq) {
        float pv = (dead || qm[q0 + qq]) ? 0.0f : __expf(sS[r][qq] - mnew);
        sP[r][qq] = (__bf16)pv;
        lsum += pv;
      }
      sL[r] = sL[r] * scale + lsum;
      sM[r] = mnew;
      sScale[r] = scale;
    }
    __syncthreads();

    // rescale O, then O += P(16x64) x repT(64x256 slice)
    float f[8];
#pragma unroll
    for (int i = 0; i < 8; ++i) f[i] = sScale[pg * 16 + i + ls * 8];
#pragma unroll
    for (int t = 0; t < 16; ++t)
#pragma unroll
      for (int i = 0; i < 8; ++i) oacc[t][i] *= f[i];

    v16bf pa[2];
#pragma unroll
    for (int k = 0; k < 2; ++k)
      pa[k] = *(const v16bf*)&sP[pg * 16 + lh][k * 32 + ls * 16];
#pragma unroll
    for (int t = 0; t < 16; ++t) {
      const int h = hg * 256 + t * 16 + lh;
#pragma unroll
      for (int k = 0; k < 2; ++k) {
        v16bf bf = *(const v16bf*)&rtg[(size_t)h * QQ + q0 + k * 32 + ls * 16];
        oacc[t] = __builtin_amdgcn_wmma_f32_16x16x32_bf16(
            false, pa[k], false, bf, (short)0, oacc[t], false, false);
      }
    }
  }

  // epilogue: O /= l, store fp32
  float inv[8];
#pragma unroll
  for (int i = 0; i < 8; ++i) inv[i] = 1.0f / sL[pg * 16 + i + ls * 8];
  float* og = Out + ((size_t)b * PP + p0 + pg * 16) * HH + hg * 256;
#pragma unroll
  for (int t = 0; t < 16; ++t)
#pragma unroll
    for (int i = 0; i < 8; ++i)
      og[(size_t)(i + ls * 8) * HH + t * 16 + lh] = oacc[t][i] * inv[i];
}

// ------------------------------- launch ------------------------------------
extern "C" void kernel_launch(void* const* d_in, const int* in_sizes, int n_in,
                              void* d_out, int out_size, void* d_ws, size_t ws_size,
                              hipStream_t stream) {
  const float*         passage  = (const float*)d_in[0];
  /* p_mask d_in[1] unused (reference ignores it) */
  const float*         question = (const float*)d_in[2];
  const unsigned char* q_mask   = (const unsigned char*)d_in[3];
  const float*         rep      = (const float*)d_in[4];
  const float*         U_w      = (const float*)d_in[5];
  const float*         D_param  = (const float*)d_in[6];
  float* out = (float*)d_out;

  char* ws = (char*)d_ws;
  __bf16* Wbf  = (__bf16*)(ws);                            //  2 MiB
  __bf16* Upbf = (__bf16*)(ws + (size_t)(2)  * (1 << 20)); // 32 MiB
  __bf16* Uqbf = (__bf16*)(ws + (size_t)(34) * (1 << 20)); // 32 MiB
  __bf16* RepT = (__bf16*)(ws + (size_t)(66) * (1 << 20)); // 32 MiB

  // 1) U_w -> bf16
  cvt_kernel<<<(HH * DI) / (256 * 4), 256, 0, stream>>>(U_w, Wbf, HH * DI);
  // 2) rep -> repT bf16
  trcvt_kernel<<<dim3(HH / 32, QQ / 32, BB), 256, 0, stream>>>(rep, RepT);
  // 3) projections (rows = B*P = B*Q = 16384)
  proj_kernel<<<dim3(HH / 128, (BB * PP) / 128), 256, 0, stream>>>(
      passage, Wbf, D_param, Upbf, 0);
  proj_kernel<<<dim3(HH / 128, (BB * QQ) / 128), 256, 0, stream>>>(
      question, Wbf, D_param, Uqbf, 1);
  // 4) flash attention
  attn_kernel<<<dim3(PP / 64, BB), 512, 0, stream>>>(Upbf, Uqbf, RepT, q_mask, out);
  (void)in_sizes; (void)n_in; (void)out_size; (void)ws_size;
}
